// ReprogrammingLayer_71914932404797
// MI455X (gfx1250) — matrix operand order repaired and checked by reading, not compiled, not run on hardware
//
#include <hip/hip_runtime.h>

typedef __attribute__((ext_vector_type(16))) _Float16 v16h;
typedef __attribute__((ext_vector_type(8)))  _Float16 v8h;
typedef __attribute__((ext_vector_type(8)))  float    v8f;

#define WMMA16(A, B, C) \
    __builtin_amdgcn_wmma_f32_16x16x32_f16(false, (A), false, (B), (short)0, (C), false, false)

// CDNA5 async global->LDS copy, 16B per lane, tracked by ASYNCcnt.
// VDST = wave-relative LDS byte address, VADDR = 64-bit global address (GV mode).
#define ASYNC_CP16(lds_off_u32, gptr)                                   \
    asm volatile("global_load_async_to_lds_b128 %0, %1, off"            \
                 :: "v"(lds_off_u32), "v"(gptr) : "memory")
#define WAIT_ASYNC_LE8() asm volatile("s_wait_asynccnt 0x8" ::: "memory")
#define WAIT_ASYNC_0()   asm volatile("s_wait_asynccnt 0x0" ::: "memory")

// ---------------------------------------------------------------------------
// Problem constants
// ---------------------------------------------------------------------------
#define BATCH   8
#define LSEQ    1024
#define D_MODEL 768
#define D_ATTN  768
#define D_LLM   4096
#define N_HEADS 12
#define HEAD_E  64
#define S_REAL  1000
#define S_PAD   1024
#define M_ROWS  (BATCH * LSEQ)   // 8192

// LDS staging geometry: 128 rows x 32 halves (64B payload) with 80B pitch
#define LDS_PITCH 80
#define TILE_BYTES (128 * LDS_PITCH)          // 10240 per buffer

enum OutMode { OUT_F16_ROW = 0, OUT_F16_TRANS = 1, OUT_F32_ROW = 2 };

// ---------------------------------------------------------------------------
// fp32 -> fp16 conversion
// ---------------------------------------------------------------------------
__global__ void cvt_f32_to_f16(const float* __restrict__ in,
                               _Float16* __restrict__ out, int n) {
    int i = blockIdx.x * blockDim.x + threadIdx.x;
    if (i < n) out[i] = (_Float16)in[i];
}

// ---------------------------------------------------------------------------
// WMMA GEMM with double-buffered async-LDS staging:
//   y[M,N] = x[M,K] @ W[N,K]^T + b[N]
// block = 128 threads (4 waves), block tile 128x128, wave tile 64x64.
// Rows >= M_real are clamp-loaded (finite garbage; callers mask them).
// ---------------------------------------------------------------------------
__global__ __launch_bounds__(128)
void gemm_wmma_f16(const _Float16* __restrict__ X, const _Float16* __restrict__ W,
                   const float* __restrict__ bias, void* __restrict__ out,
                   int M_real, int K, int ldo, int out_mode) {
    __shared__ char sLds[4 * TILE_BYTES];     // A0, A1, B0, B1 (40 KB)

    const int t    = threadIdx.x;
    const int lane = t & 31;
    const int wave = t >> 5;
    const int lr = lane & 15;   // row/col index within 16-lane group
    const int lg = lane >> 4;   // half-group 0/1

    const int m_block = blockIdx.x * 128;
    const int n_block = blockIdx.y * 128;
    const int m0 = m_block + (wave >> 1) * 64;
    const int n0 = n_block + (wave & 1) * 64;

    // Generic pointer low 32 bits == wave-relative LDS offset (flat aperture rule)
    const unsigned ldsBase = (unsigned)(unsigned long long)(void*)&sLds[0];

    // Issue the async copies for one 128x32 A tile + 128x32 B tile (8 per thread)
    auto stage = [&](int buf, int kk) {
#pragma unroll
        for (int j = 0; j < 4; ++j) {
            const int c = j * 128 + t;        // chunk 0..511
            const int row = c >> 2;           // 0..127
            const int q = c & 3;              // 16B chunk within 64B row payload
            int arow = m_block + row;
            if (arow >= M_real) arow = M_real - 1;
            const _Float16* ga = X + (size_t)arow * K + kk + q * 8;
            ASYNC_CP16(ldsBase + (unsigned)(buf * TILE_BYTES + row * LDS_PITCH + q * 16), ga);
            const _Float16* gb = W + (size_t)(n_block + row) * K + kk + q * 8;
            ASYNC_CP16(ldsBase + (unsigned)((2 + buf) * TILE_BYTES + row * LDS_PITCH + q * 16), gb);
        }
    };

    v8f acc[4][4] = {};
    const int nsteps = K >> 5;

    stage(0, 0);
    for (int step = 0; step < nsteps; ++step) {
        const int cur = step & 1;
        const bool more = (step + 1 < nsteps);
        if (more) stage(cur ^ 1, (step + 1) * 32);   // prefetch next buffer
        if (more) WAIT_ASYNC_LE8(); else WAIT_ASYNC_0();  // current buffer complete
        __syncthreads();

        const char* sA = &sLds[cur * TILE_BYTES];
        const char* sB = &sLds[(2 + cur) * TILE_BYTES];

        v16h A[4], B[4];
#pragma unroll
        for (int mt = 0; mt < 4; ++mt) {
            const _Float16* p = (const _Float16*)(sA + ((wave >> 1) * 64 + mt * 16 + lr) * LDS_PITCH + lg * 16);
            union { v16h v; v8h h[2]; } u;
            u.h[0] = *(const v8h*)(p);        // k = lg*8 + 0..7
            u.h[1] = *(const v8h*)(p + 16);   // k = lg*8 + 16..23
            A[mt] = u.v;
        }
#pragma unroll
        for (int nt = 0; nt < 4; ++nt) {
            const _Float16* p = (const _Float16*)(sB + ((wave & 1) * 64 + nt * 16 + lr) * LDS_PITCH + lg * 16);
            union { v16h v; v8h h[2]; } u;
            u.h[0] = *(const v8h*)(p);        // k = lg*8 + 0..7
            u.h[1] = *(const v8h*)(p + 16);   // k = lg*8 + 16..23
            B[nt] = u.v;
        }
#pragma unroll
        for (int mt = 0; mt < 4; ++mt)
#pragma unroll
            for (int nt = 0; nt < 4; ++nt)
                acc[mt][nt] = WMMA16(A[mt], B[nt], acc[mt][nt]);

        __syncthreads();   // everyone done reading before this buffer is restaged
    }

    // Epilogue: D layout — lane gives n = lr, m = r + 8*lg per accumulator elem r
#pragma unroll
    for (int nt = 0; nt < 4; ++nt) {
        const int col = n0 + nt * 16 + lr;
        const float bv = bias[col];
#pragma unroll
        for (int mt = 0; mt < 4; ++mt) {
            v8f c = acc[mt][nt];
#pragma unroll
            for (int r = 0; r < 8; ++r) {
                const int row = m0 + mt * 16 + r + 8 * lg;
                const float val = c[r] + bv;
                if (out_mode == OUT_F32_ROW) {
                    ((float*)out)[(size_t)row * ldo + col] = val;
                } else if (out_mode == OUT_F16_ROW) {
                    ((_Float16*)out)[(size_t)row * ldo + col] = (_Float16)val;
                } else { // OUT_F16_TRANS
                    ((_Float16*)out)[(size_t)col * ldo + row] = (_Float16)val;
                }
            }
        }
    }
}

// ---------------------------------------------------------------------------
// Attention: one wave per (b, h, 16-row L tile).
//   scores = (Q Kt) * 0.125, masked s>=1000 -> softmax -> P @ V
//   Q  : (8192, 768) f16 row-major
//   Kt : (1024, 768) f16 row-major   (K matrix; rows >= 1000 are garbage)
//   Vt : (768, 1024) f16 row-major   (V transposed: Vt[c][s])
//   Rep: (8192, 768) f16 row-major
// ---------------------------------------------------------------------------
__global__ __launch_bounds__(32)
void attn_wmma(const _Float16* __restrict__ Q, const _Float16* __restrict__ Kt,
               const _Float16* __restrict__ Vt, _Float16* __restrict__ Rep) {
    __shared__ _Float16 sP[16 * S_PAD];   // scores, then probs (in place)
    __shared__ float sSum[16];

    const int lane = threadIdx.x & 31;
    const int lr = lane & 15;
    const int lg = lane >> 4;

    const int wg = blockIdx.x;             // ((b*12)+h)*64 + lt
    const int lt = wg & 63;
    const int h  = (wg >> 6) % N_HEADS;
    const int b  = wg / (64 * N_HEADS);
    const int l0 = b * LSEQ + lt * 16;     // base row in flattened (B*L)

    // Q tile as A operand, K-dim = head_dim = 64 -> two 16x32 operands
    v16h A0, A1;
    {
        const _Float16* qp = Q + (size_t)(l0 + lr) * D_ATTN + h * HEAD_E;
        union { v16h v; v8h p[2]; } u;
        u.p[0] = *(const v8h*)(qp + lg * 16);
        u.p[1] = *(const v8h*)(qp + lg * 16 + 8);
        A0 = u.v;
        u.p[0] = *(const v8h*)(qp + 32 + lg * 16);
        u.p[1] = *(const v8h*)(qp + 32 + lg * 16 + 8);
        A1 = u.v;
    }

    // ---- scores: 64 s-tiles of 16 ----
    for (int st = 0; st < 64; ++st) {
        const int s_base = st * 16;
        const _Float16* kp = Kt + (size_t)(s_base + lr) * D_ATTN + h * HEAD_E + lg * 16;
        v16h B0 = *(const v16h*)(kp);        // k (=e) in [lg*16, lg*16+16)
        v16h B1 = *(const v16h*)(kp + 32);   // k in [32+lg*16, ...)
        v8f c = {};
        c = WMMA16(A0, B0, c);
        c = WMMA16(A1, B1, c);
        const int s = s_base + lr;           // D-layout: n = lr
        const bool valid = (s < S_REAL);
#pragma unroll
        for (int r = 0; r < 8; ++r) {
            const int m = r + 8 * lg;
            float sc = valid ? c[r] * 0.125f : -65504.0f;
            sP[m * S_PAD + s] = (_Float16)sc;
        }
    }
    __syncthreads();

    // ---- softmax: lane owns (row = lr, half = lg covering 512 cols) ----
    float mx = -3.0e38f;
    for (int s = lg * 512; s < lg * 512 + 512; ++s)
        mx = fmaxf(mx, (float)sP[lr * S_PAD + s]);
    mx = fmaxf(mx, __shfl_xor(mx, 16, 32));

    float sum = 0.0f;
    for (int s = lg * 512; s < lg * 512 + 512; ++s) {
        float p = __expf((float)sP[lr * S_PAD + s] - mx);  // masked cols -> 0
        sum += p;
        sP[lr * S_PAD + s] = (_Float16)p;
    }
    sum += __shfl_xor(sum, 16, 32);
    if (lg == 0) sSum[lr] = sum;
    __syncthreads();

    // ---- PV: O[16 x 64] = P[16 x 1024] @ V[1024 x 64] ----
    v8f O[4] = {};
    for (int sk = 0; sk < 32; ++sk) {
        union { v16h v; v8h p[2]; } u;
        const _Float16* pp = &sP[lr * S_PAD + sk * 32 + lg * 16];
        u.p[0] = *(const v8h*)(pp);
        u.p[1] = *(const v8h*)(pp + 8);
        v16h Ap = u.v;
#pragma unroll
        for (int et = 0; et < 4; ++et) {
            const _Float16* vp = Vt + (size_t)(h * HEAD_E + et * 16 + lr) * S_PAD
                                    + sk * 32 + lg * 16;
            v16h Bv = *(const v16h*)(vp);
            O[et] = WMMA16(Ap, Bv, O[et]);
        }
    }

    // ---- epilogue: divide by row sums, store f16 ----
    float inv[8];
#pragma unroll
    for (int r = 0; r < 8; ++r) inv[r] = 1.0f / sSum[r + 8 * lg];
#pragma unroll
    for (int et = 0; et < 4; ++et) {
#pragma unroll
        for (int r = 0; r < 8; ++r) {
            const int m = r + 8 * lg;
            const float val = O[et][r] * inv[r];
            Rep[(size_t)(l0 + m) * D_ATTN + h * HEAD_E + et * 16 + lr] = (_Float16)val;
        }
    }
}

// ---------------------------------------------------------------------------
// Host launcher
// ---------------------------------------------------------------------------
extern "C" void kernel_launch(void* const* d_in, const int* in_sizes, int n_in,
                              void* d_out, int out_size, void* d_ws, size_t ws_size,
                              hipStream_t stream) {
    (void)in_sizes; (void)n_in; (void)out_size; (void)ws_size;

    const float* target = (const float*)d_in[0];   // (8,1024,768)
    const float* source = (const float*)d_in[1];   // (1000,4096)
    const float* value  = (const float*)d_in[2];   // (1000,4096)
    const float* Wq = (const float*)d_in[3];
    const float* bq = (const float*)d_in[4];
    const float* Wk = (const float*)d_in[5];
    const float* bk = (const float*)d_in[6];
    const float* Wv = (const float*)d_in[7];
    const float* bv = (const float*)d_in[8];
    const float* Wo = (const float*)d_in[9];
    const float* bo = (const float*)d_in[10];
    float* out = (float*)d_out;

    // workspace carve-up (f16 buffers), 256B aligned
    char* ws = (char*)d_ws;
    size_t off = 0;
    auto carve = [&](size_t elems) -> _Float16* {
        _Float16* p = (_Float16*)(ws + off);
        off = (off + elems * sizeof(_Float16) + 255) & ~(size_t)255;
        return p;
    };
    _Float16* Xt_h  = carve((size_t)M_ROWS * D_MODEL);   // target f16
    _Float16* Wq_h  = carve((size_t)D_ATTN * D_MODEL);
    _Float16* Src_h = carve((size_t)S_REAL * D_LLM);
    _Float16* Val_h = carve((size_t)S_REAL * D_LLM);
    _Float16* Wk_h  = carve((size_t)D_ATTN * D_LLM);
    _Float16* Wv_h  = carve((size_t)D_ATTN * D_LLM);
    _Float16* Wo_h  = carve((size_t)D_LLM * D_ATTN);
    _Float16* Q_h   = carve((size_t)M_ROWS * D_ATTN);
    _Float16* K_h   = carve((size_t)S_PAD * D_ATTN);     // padded rows
    _Float16* Vt_h  = carve((size_t)D_ATTN * S_PAD);     // V transposed
    _Float16* Rep_h = carve((size_t)M_ROWS * D_ATTN);

    auto cvt = [&](const float* src, _Float16* dst, int n) {
        cvt_f32_to_f16<<<(n + 255) / 256, 256, 0, stream>>>(src, dst, n);
    };
    cvt(target, Xt_h,  M_ROWS * D_MODEL);
    cvt(Wq,     Wq_h,  D_ATTN * D_MODEL);
    cvt(source, Src_h, S_REAL * D_LLM);
    cvt(value,  Val_h, S_REAL * D_LLM);
    cvt(Wk,     Wk_h,  D_ATTN * D_LLM);
    cvt(Wv,     Wv_h,  D_ATTN * D_LLM);
    cvt(Wo,     Wo_h,  D_LLM * D_ATTN);

    // Q = target @ Wq^T + bq        (8192 x 768), f16 row-major
    gemm_wmma_f16<<<dim3(M_ROWS / 128, D_ATTN / 128), 128, 0, stream>>>(
        Xt_h, Wq_h, bq, Q_h, M_ROWS, D_MODEL, D_ATTN, OUT_F16_ROW);

    // K = source @ Wk^T + bk        (1024(pad) x 768), f16 row-major
    gemm_wmma_f16<<<dim3(S_PAD / 128, D_ATTN / 128), 128, 0, stream>>>(
        Src_h, Wk_h, bk, K_h, S_REAL, D_LLM, D_ATTN, OUT_F16_ROW);

    // V = value @ Wv^T + bv, stored transposed as Vt (768 x 1024)
    gemm_wmma_f16<<<dim3(S_PAD / 128, D_ATTN / 128), 128, 0, stream>>>(
        Val_h, Wv_h, bv, Vt_h, S_REAL, D_LLM, S_PAD, OUT_F16_TRANS);

    // Attention -> Rep (8192 x 768) f16
    attn_wmma<<<BATCH * N_HEADS * 64, 32, 0, stream>>>(Q_h, K_h, Vt_h, Rep_h);

    // out = Rep @ Wo^T + bo         (8192 x 4096), fp32
    gemm_wmma_f16<<<dim3(M_ROWS / 128, D_LLM / 128), 128, 0, stream>>>(
        Rep_h, Wo_h, bo, out, M_ROWS, D_ATTN, D_LLM, OUT_F32_ROW);
}
